// IDST_1133871366818
// MI455X (gfx1250) — matrix operand order repaired and checked by reading, not compiled, not run on hardware
//
#include <hip/hip_runtime.h>
#include <hip/hip_bf16.h>
#include <math.h>

// ---------------------------------------------------------------------------
// IDST (DST-III) as dense GEMM: Y[M,N] = X[M,K] * C[K,N]
//   C[k][j] = 2*a_k*sin(pi*(k+1)*(2j+1)/(2N)), a_{K-1}=0.5
// M = N = K = 4096.  B stored transposed (Bt[n][k]) in bf16 hi/lo planes;
// X pre-split into bf16 hi/lo planes; GEMM uses v_wmma_f32_16x16x32_bf16
// with the 3-term hi/lo split (AhBh + AhBl + AlBh) for near-fp32 accuracy.
// LDS staging via CDNA5 async global->LDS (GLOBAL_LOAD_ASYNC_TO_LDS_B128).
// ---------------------------------------------------------------------------

typedef __attribute__((ext_vector_type(16))) __bf16 v16bf;
typedef __attribute__((ext_vector_type(8)))  __bf16 v8bf;
typedef __attribute__((ext_vector_type(8)))  float  v8f;
typedef int v4i_vs __attribute__((vector_size(16)));   // matches builtin param type

#define MDIM 4096
#define NDIM 4096
#define KDIM 4096

#define BM 128
#define BN 128
#define BK 32
#define LDE 40   // padded LDS row stride in bf16 elements (80 B = 5*16 B)

#if defined(__has_builtin)
#  if __has_builtin(__builtin_amdgcn_global_load_async_to_lds_b128)
#    define ASYNC_LDS 1
#  endif
#endif
#ifndef ASYNC_LDS
#  define ASYNC_LDS 0
#endif

static __device__ __forceinline__ unsigned short f32_to_bf16_rne(float f) {
    unsigned int u = __float_as_uint(f);
    unsigned int r = u + 0x7FFFu + ((u >> 16) & 1u);
    return (unsigned short)(r >> 16);
}
static __device__ __forceinline__ float bf16_bits_to_f32(unsigned short b) {
    return __uint_as_float(((unsigned int)b) << 16);
}

// 16-byte global -> LDS copy. Async path uses GLOBAL_LOAD_ASYNC_TO_LDS_B128
// (ASYNCcnt-tracked); fallback is a synchronous load+store through VGPRs.
static __device__ __forceinline__ void stage16(const unsigned short* g, unsigned short* l) {
#if ASYNC_LDS
    __builtin_amdgcn_global_load_async_to_lds_b128(
        (__attribute__((address_space(1))) v4i_vs*)g,
        (__attribute__((address_space(3))) v4i_vs*)l, 0, 0);
#else
    *(uint4*)l = *(const uint4*)g;
#endif
}

static __device__ __forceinline__ void stage_wait() {
#if ASYNC_LDS
#  if __has_builtin(__builtin_amdgcn_s_wait_asynccnt)
    __builtin_amdgcn_s_wait_asynccnt(0);
#  else
    asm volatile("s_wait_asynccnt 0x0" ::: "memory");
#  endif
#endif
}

// ---------------------------------------------------------------------------
// Build Bt[n][k] = C[k][n] in bf16 hi (+ optional lo) planes.
// Phase integer reduced mod 4N = 16384 (power of two) => exact reduction.
// ---------------------------------------------------------------------------
__global__ __launch_bounds__(256) void build_Bt(unsigned short* __restrict__ bhi,
                                                unsigned short* __restrict__ blo,
                                                int split) {
    const float PI_OVER_8192 = 3.83495196e-4f;  // pi / 8192
    const long long total = (long long)NDIM * KDIM;
    for (long long idx = (long long)blockIdx.x * blockDim.x + threadIdx.x;
         idx < total; idx += (long long)gridDim.x * blockDim.x) {
        int n = (int)(idx >> 12);      // row of Bt  (output column j)
        int k = (int)(idx & 4095);     // reduction index
        unsigned int ph = ((unsigned int)(k + 1) * (unsigned int)(2 * n + 1)) & 16383u;
        float s = sinf((float)ph * PI_OVER_8192);
        float val = (k == KDIM - 1 ? 1.0f : 2.0f) * s;
        unsigned short h = f32_to_bf16_rne(val);
        bhi[idx] = h;
        if (split) blo[idx] = f32_to_bf16_rne(val - bf16_bits_to_f32(h));
    }
}

// ---------------------------------------------------------------------------
// One-time X -> bf16 hi/lo planes (row-major [m][k], same layout as X).
// ---------------------------------------------------------------------------
__global__ __launch_bounds__(256) void split_X(const float* __restrict__ x,
                                               unsigned short* __restrict__ xhi,
                                               unsigned short* __restrict__ xlo) {
    const long long total4 = (long long)MDIM * KDIM / 4;
    for (long long i = (long long)blockIdx.x * blockDim.x + threadIdx.x;
         i < total4; i += (long long)gridDim.x * blockDim.x) {
        float4 v = ((const float4*)x)[i];
        unsigned short h0 = f32_to_bf16_rne(v.x);
        unsigned short h1 = f32_to_bf16_rne(v.y);
        unsigned short h2 = f32_to_bf16_rne(v.z);
        unsigned short h3 = f32_to_bf16_rne(v.w);
        uint2 hp;
        hp.x = (unsigned)h0 | ((unsigned)h1 << 16);
        hp.y = (unsigned)h2 | ((unsigned)h3 << 16);
        ((uint2*)xhi)[i] = hp;
        unsigned short l0 = f32_to_bf16_rne(v.x - bf16_bits_to_f32(h0));
        unsigned short l1 = f32_to_bf16_rne(v.y - bf16_bits_to_f32(h1));
        unsigned short l2 = f32_to_bf16_rne(v.z - bf16_bits_to_f32(h2));
        unsigned short l3 = f32_to_bf16_rne(v.w - bf16_bits_to_f32(h3));
        uint2 lp;
        lp.x = (unsigned)l0 | ((unsigned)l1 << 16);
        lp.y = (unsigned)l2 | ((unsigned)l3 << 16);
        ((uint2*)xlo)[i] = lp;
    }
}

// ---------------------------------------------------------------------------
// Plane-mode WMMA GEMM: all four tiles (Xhi/Xlo/Bhi/Blo) are pure bf16
// copies, staged with async global->LDS. 256 threads = 8 waves (wave32).
// Block tile 128x128, BK=32; wave grid 2x4; wave computes 4x2 WMMA tiles.
// ---------------------------------------------------------------------------
__global__ __launch_bounds__(256) void idst_gemm_async(const unsigned short* __restrict__ Xhi,
                                                       const unsigned short* __restrict__ Xlo,
                                                       const unsigned short* __restrict__ Bhi,
                                                       const unsigned short* __restrict__ Blo,
                                                       float* __restrict__ Y) {
    __shared__ unsigned short sXh[BM * LDE];
    __shared__ unsigned short sXl[BM * LDE];
    __shared__ unsigned short sBh[BN * LDE];
    __shared__ unsigned short sBl[BN * LDE];

    const int tid  = threadIdx.x;
    const int wid  = tid >> 5;
    const int lane = tid & 31;
    const int wm   = wid & 1;        // 2 waves along M
    const int wn   = wid >> 1;       // 4 waves along N
    const int lm   = lane & 15;
    const int lh   = lane >> 4;

    const long long rowBase = (long long)blockIdx.y * BM;
    const int       colBase = blockIdx.x * BN;

    v8f acc[4][2];
    const v8f vzero = {0.f, 0.f, 0.f, 0.f, 0.f, 0.f, 0.f, 0.f};
    for (int i = 0; i < 4; ++i)
        for (int j = 0; j < 2; ++j) acc[i][j] = vzero;

    for (int kb = 0; kb < KDIM; kb += BK) {
        // ---- stage 4 bf16 planes, 16-B chunks: 2 chunks/thread/plane ----
        for (int it = 0; it < 2; ++it) {
            int idx = tid + it * 256;           // 0..511
            int r   = idx >> 2;                 // 0..127
            int c8  = (idx & 3) << 3;           // 0,8,16,24  (bf16 elements)
            long long gx = (rowBase + r) * KDIM + kb + c8;
            long long gb = (long long)(colBase + r) * KDIM + kb + c8;
            stage16(Xhi + gx, &sXh[r * LDE + c8]);
            stage16(Xlo + gx, &sXl[r * LDE + c8]);
            stage16(Bhi + gb, &sBh[r * LDE + c8]);
            stage16(Blo + gb, &sBl[r * LDE + c8]);
        }
        stage_wait();
        __syncthreads();

        // ---- gather fragments per ISA VGPR layouts ----
        v16bf afh[4], afl[4], bth[2], btl[2];
        for (int mt = 0; mt < 4; ++mt) {
            int r = wm * 64 + mt * 16 + lm;
            v8bf c0 = *(const v8bf*)(&sXh[r * LDE + 8 * lh]);
            v8bf c1 = *(const v8bf*)(&sXh[r * LDE + 16 + 8 * lh]);
            afh[mt] = __builtin_shufflevector(c0, c1, 0, 1, 2, 3, 4, 5, 6, 7,
                                              8, 9, 10, 11, 12, 13, 14, 15);
            v8bf d0 = *(const v8bf*)(&sXl[r * LDE + 8 * lh]);
            v8bf d1 = *(const v8bf*)(&sXl[r * LDE + 16 + 8 * lh]);
            afl[mt] = __builtin_shufflevector(d0, d1, 0, 1, 2, 3, 4, 5, 6, 7,
                                              8, 9, 10, 11, 12, 13, 14, 15);
        }
        for (int nt = 0; nt < 2; ++nt) {
            int c = wn * 32 + nt * 16 + lm;
            v8bf c0 = *(const v8bf*)(&sBh[c * LDE + 8 * lh]);
            v8bf c1 = *(const v8bf*)(&sBh[c * LDE + 16 + 8 * lh]);
            bth[nt] = __builtin_shufflevector(c0, c1, 0, 1, 2, 3, 4, 5, 6, 7,
                                              8, 9, 10, 11, 12, 13, 14, 15);
            v8bf d0 = *(const v8bf*)(&sBl[c * LDE + 8 * lh]);
            v8bf d1 = *(const v8bf*)(&sBl[c * LDE + 16 + 8 * lh]);
            btl[nt] = __builtin_shufflevector(d0, d1, 0, 1, 2, 3, 4, 5, 6, 7,
                                              8, 9, 10, 11, 12, 13, 14, 15);
        }

        for (int mt = 0; mt < 4; ++mt)
            for (int nt = 0; nt < 2; ++nt) {
                acc[mt][nt] = __builtin_amdgcn_wmma_f32_16x16x32_bf16(
                    false, afh[mt], false, bth[nt], (short)0, acc[mt][nt], false, false);
                acc[mt][nt] = __builtin_amdgcn_wmma_f32_16x16x32_bf16(
                    false, afh[mt], false, btl[nt], (short)0, acc[mt][nt], false, false);
                acc[mt][nt] = __builtin_amdgcn_wmma_f32_16x16x32_bf16(
                    false, afl[mt], false, bth[nt], (short)0, acc[mt][nt], false, false);
            }
        __syncthreads();
    }

    for (int mt = 0; mt < 4; ++mt)
        for (int nt = 0; nt < 2; ++nt) {
            int col = colBase + wn * 32 + nt * 16 + lm;
            long long mr = rowBase + wm * 64 + mt * 16 + lh * 8;
            for (int r = 0; r < 8; ++r)
                Y[(mr + r) * NDIM + col] = acc[mt][nt][r];
        }
}

// ---------------------------------------------------------------------------
// On-the-fly conversion GEMM (used when ws holds only the B planes).
// ---------------------------------------------------------------------------
template <int SPLIT>
__global__ __launch_bounds__(256) void idst_gemm(const float* __restrict__ X,
                                                 const unsigned short* __restrict__ Bhi,
                                                 const unsigned short* __restrict__ Blo,
                                                 float* __restrict__ Y) {
    __shared__ unsigned short sXh[BM * LDE];
    __shared__ unsigned short sXl[SPLIT ? BM * LDE : 1];
    __shared__ unsigned short sBh[BN * LDE];
    __shared__ unsigned short sBl[SPLIT ? BN * LDE : 1];

    const int tid  = threadIdx.x;
    const int wid  = tid >> 5;
    const int lane = tid & 31;
    const int wm   = wid & 1;
    const int wn   = wid >> 1;
    const int lm   = lane & 15;
    const int lh   = lane >> 4;

    const long long rowBase = (long long)blockIdx.y * BM;
    const int       colBase = blockIdx.x * BN;

    v8f acc[4][2];
    const v8f vzero = {0.f, 0.f, 0.f, 0.f, 0.f, 0.f, 0.f, 0.f};
    for (int i = 0; i < 4; ++i)
        for (int j = 0; j < 2; ++j) acc[i][j] = vzero;

    for (int kb = 0; kb < KDIM; kb += BK) {
        for (int it = 0; it < 4; ++it) {
            int idx = tid + it * 256;
            int r   = idx >> 3;
            int c4  = (idx & 7) << 2;
            const float4 v = *(const float4*)(X + (rowBase + r) * KDIM + kb + c4);
            unsigned short h0 = f32_to_bf16_rne(v.x);
            unsigned short h1 = f32_to_bf16_rne(v.y);
            unsigned short h2 = f32_to_bf16_rne(v.z);
            unsigned short h3 = f32_to_bf16_rne(v.w);
            uint2 hp;
            hp.x = (unsigned)h0 | ((unsigned)h1 << 16);
            hp.y = (unsigned)h2 | ((unsigned)h3 << 16);
            *(uint2*)(&sXh[r * LDE + c4]) = hp;
            if (SPLIT) {
                unsigned short l0 = f32_to_bf16_rne(v.x - bf16_bits_to_f32(h0));
                unsigned short l1 = f32_to_bf16_rne(v.y - bf16_bits_to_f32(h1));
                unsigned short l2 = f32_to_bf16_rne(v.z - bf16_bits_to_f32(h2));
                unsigned short l3 = f32_to_bf16_rne(v.w - bf16_bits_to_f32(h3));
                uint2 lp;
                lp.x = (unsigned)l0 | ((unsigned)l1 << 16);
                lp.y = (unsigned)l2 | ((unsigned)l3 << 16);
                *(uint2*)(&sXl[r * LDE + c4]) = lp;
            }
        }
        for (int it = 0; it < 2; ++it) {
            int idx = tid + it * 256;
            int n   = idx >> 2;
            int c8  = (idx & 3) << 3;
            long long g = (long long)(colBase + n) * KDIM + kb + c8;
            *(uint4*)(&sBh[n * LDE + c8]) = *(const uint4*)(Bhi + g);
            if (SPLIT) *(uint4*)(&sBl[n * LDE + c8]) = *(const uint4*)(Blo + g);
        }
        __syncthreads();

        v16bf afh[4], afl[4], bth[2], btl[2];
        for (int mt = 0; mt < 4; ++mt) {
            int r = wm * 64 + mt * 16 + lm;
            v8bf c0 = *(const v8bf*)(&sXh[r * LDE + 8 * lh]);
            v8bf c1 = *(const v8bf*)(&sXh[r * LDE + 16 + 8 * lh]);
            afh[mt] = __builtin_shufflevector(c0, c1, 0, 1, 2, 3, 4, 5, 6, 7,
                                              8, 9, 10, 11, 12, 13, 14, 15);
            if (SPLIT) {
                v8bf d0 = *(const v8bf*)(&sXl[r * LDE + 8 * lh]);
                v8bf d1 = *(const v8bf*)(&sXl[r * LDE + 16 + 8 * lh]);
                afl[mt] = __builtin_shufflevector(d0, d1, 0, 1, 2, 3, 4, 5, 6, 7,
                                                  8, 9, 10, 11, 12, 13, 14, 15);
            }
        }
        for (int nt = 0; nt < 2; ++nt) {
            int c = wn * 32 + nt * 16 + lm;
            v8bf c0 = *(const v8bf*)(&sBh[c * LDE + 8 * lh]);
            v8bf c1 = *(const v8bf*)(&sBh[c * LDE + 16 + 8 * lh]);
            bth[nt] = __builtin_shufflevector(c0, c1, 0, 1, 2, 3, 4, 5, 6, 7,
                                              8, 9, 10, 11, 12, 13, 14, 15);
            if (SPLIT) {
                v8bf d0 = *(const v8bf*)(&sBl[c * LDE + 8 * lh]);
                v8bf d1 = *(const v8bf*)(&sBl[c * LDE + 16 + 8 * lh]);
                btl[nt] = __builtin_shufflevector(d0, d1, 0, 1, 2, 3, 4, 5, 6, 7,
                                                  8, 9, 10, 11, 12, 13, 14, 15);
            }
        }

        for (int mt = 0; mt < 4; ++mt)
            for (int nt = 0; nt < 2; ++nt) {
                acc[mt][nt] = __builtin_amdgcn_wmma_f32_16x16x32_bf16(
                    false, afh[mt], false, bth[nt], (short)0, acc[mt][nt], false, false);
                if (SPLIT) {
                    acc[mt][nt] = __builtin_amdgcn_wmma_f32_16x16x32_bf16(
                        false, afh[mt], false, btl[nt], (short)0, acc[mt][nt], false, false);
                    acc[mt][nt] = __builtin_amdgcn_wmma_f32_16x16x32_bf16(
                        false, afl[mt], false, bth[nt], (short)0, acc[mt][nt], false, false);
                }
            }
        __syncthreads();
    }

    for (int mt = 0; mt < 4; ++mt)
        for (int nt = 0; nt < 2; ++nt) {
            int col = colBase + wn * 32 + nt * 16 + lm;
            long long mr = rowBase + wm * 64 + mt * 16 + lh * 8;
            for (int r = 0; r < 8; ++r)
                Y[(mr + r) * NDIM + col] = acc[mt][nt][r];
        }
}

// ---------------------------------------------------------------------------
// Naive fallback (only if workspace is too small for even one bf16 plane).
// ---------------------------------------------------------------------------
__global__ __launch_bounds__(256) void idst_naive(const float* __restrict__ x,
                                                  float* __restrict__ y) {
    long long idx = (long long)blockIdx.x * 256 + threadIdx.x;
    if (idx >= (long long)MDIM * NDIM) return;
    int m = (int)(idx >> 12);
    int n = (int)(idx & 4095);
    const float PI_OVER_8192 = 3.83495196e-4f;
    float a = 0.0f;
    for (int k = 0; k < KDIM; ++k) {
        unsigned int ph = ((unsigned int)(k + 1) * (unsigned int)(2 * n + 1)) & 16383u;
        float s = __sinf((float)ph * PI_OVER_8192);
        float w = (k == KDIM - 1 ? 1.0f : 2.0f) * s;
        a = fmaf(x[(long long)m * KDIM + k], w, a);
    }
    y[idx] = a;
}

extern "C" void kernel_launch(void* const* d_in, const int* in_sizes, int n_in,
                              void* d_out, int out_size, void* d_ws, size_t ws_size,
                              hipStream_t stream) {
    const float* x = (const float*)d_in[0];
    // d_in[1] (expk) unused: twiddles are regenerated exactly (mod-4N integer phase).
    float* y = (float*)d_out;

    const size_t planeElems = (size_t)NDIM * KDIM;
    const size_t plane = planeElems * sizeof(unsigned short);  // 32 MB
    dim3 grid(NDIM / BN, MDIM / BM);

    if (ws_size >= 4 * plane) {
        // Full plane mode: B hi/lo + X hi/lo, async-staged GEMM.
        unsigned short* bhi = (unsigned short*)d_ws;
        unsigned short* blo = bhi + planeElems;
        unsigned short* xhi = blo + planeElems;
        unsigned short* xlo = xhi + planeElems;
        build_Bt<<<8192, 256, 0, stream>>>(bhi, blo, 1);
        split_X<<<4096, 256, 0, stream>>>(x, xhi, xlo);
        idst_gemm_async<<<grid, 256, 0, stream>>>(xhi, xlo, bhi, blo, y);
    } else if (ws_size >= 2 * plane) {
        unsigned short* bhi = (unsigned short*)d_ws;
        unsigned short* blo = bhi + planeElems;
        build_Bt<<<8192, 256, 0, stream>>>(bhi, blo, 1);
        idst_gemm<1><<<grid, 256, 0, stream>>>(x, bhi, blo, y);
    } else if (ws_size >= plane) {
        unsigned short* bhi = (unsigned short*)d_ws;
        build_Bt<<<8192, 256, 0, stream>>>(bhi, bhi, 0);
        idst_gemm<0><<<grid, 256, 0, stream>>>(x, bhi, bhi, y);
    } else {
        long long tot = (long long)MDIM * NDIM;
        idst_naive<<<(unsigned)((tot + 255) / 256), 256, 0, stream>>>(x, y);
    }
}
